// MultiHeadGAT_18356690223827
// MI455X (gfx1250) — compile-verified, matrix operand order checked by reading, not run on hardware
//
#include <hip/hip_runtime.h>
#include <hip/hip_bf16.h>
#include <math.h>

typedef __bf16 bf16_t;
typedef __attribute__((ext_vector_type(16))) __bf16 v16bf;
typedef __attribute__((ext_vector_type(8)))  __bf16 v8bf;
typedef __attribute__((ext_vector_type(8)))  float  v8f;

#define NEG_SLOPE 0.2f

__device__ __forceinline__ float leaky(float x) { return x >= 0.f ? x : NEG_SLOPE * x; }

// Build a bf16 A/B fragment (16x16x32 layout) from two contiguous 8-element runs.
__device__ __forceinline__ v16bf pack_frag_f32(float4 x0, float4 x1, float4 y0, float4 y1) {
  v16bf a;
  a[0]=(bf16_t)x0.x;  a[1]=(bf16_t)x0.y;  a[2]=(bf16_t)x0.z;  a[3]=(bf16_t)x0.w;
  a[4]=(bf16_t)x1.x;  a[5]=(bf16_t)x1.y;  a[6]=(bf16_t)x1.z;  a[7]=(bf16_t)x1.w;
  a[8]=(bf16_t)y0.x;  a[9]=(bf16_t)y0.y;  a[10]=(bf16_t)y0.z; a[11]=(bf16_t)y0.w;
  a[12]=(bf16_t)y1.x; a[13]=(bf16_t)y1.y; a[14]=(bf16_t)y1.z; a[15]=(bf16_t)y1.w;
  return a;
}

__global__ __launch_bounds__(512)
void gat_fused_kernel(const float* __restrict__ h,
                      const float* __restrict__ W,
                      const float* __restrict__ att_a,
                      float* __restrict__ out)
{
  constexpr int N = 512, K = 128, NH = 8, D = 64;
  extern __shared__ char smem_raw[];
  bf16_t* Wt   = (bf16_t*)smem_raw;                        // [D][K]  bf16, K-contig
  bf16_t* HpT  = (bf16_t*)(smem_raw + D * K * 2);          // [D][N]  bf16, m-contig
  float*  esrc = (float*)(smem_raw + D * K * 2 + D * N * 2); // [N]
  float*  edst = esrc + N;                                   // [N]

  const int bh   = blockIdx.x;
  const int b    = bh >> 3;
  const int head = bh & 7;
  const int tid  = threadIdx.x;
  const int wave = tid >> 5;
  const int lane = tid & 31;
  const int half = lane >> 4;
  const int l16  = lane & 15;
  const int rowbase = wave * 32;

  const float* hB = h + (size_t)b * N * K;

  // ---- Stage W head-slice transposed into LDS (coalesced global reads).
  for (int i = tid; i < D * K; i += 512) {
    int k = i >> 6, d = i & 63;
    Wt[d * K + k] = (bf16_t)W[k * (NH * D) + head * D + d];
  }
  __syncthreads();

  // Per-lane slices of the attention vectors a_src / a_dst.
  float asrc_l[4], adst_l[4];
  #pragma unroll
  for (int nt = 0; nt < 4; ++nt) {
    asrc_l[nt] = att_a[head * (2 * D) + nt * 16 + l16];
    adst_l[nt] = att_a[head * (2 * D) + D + nt * 16 + l16];
  }

  // ---- GEMM1: Hp[rowbase..rowbase+31, 0..63] = h[b] @ W_head  (bf16 WMMA, f32 acc)
  v8f acc[2][4];
  #pragma unroll
  for (int mt = 0; mt < 2; ++mt)
    #pragma unroll
    for (int nt = 0; nt < 4; ++nt)
      #pragma unroll
      for (int i = 0; i < 8; ++i) acc[mt][nt][i] = 0.f;

  #pragma unroll
  for (int kk = 0; kk < K; kk += 32) {
    v16bf afr[2];
    #pragma unroll
    for (int mt = 0; mt < 2; ++mt) {
      const float* p = hB + (size_t)(rowbase + mt * 16 + l16) * K + kk + half * 8;
      afr[mt] = pack_frag_f32(*(const float4*)(p),      *(const float4*)(p + 4),
                              *(const float4*)(p + 16), *(const float4*)(p + 20));
    }
    v16bf bfr[4];
    #pragma unroll
    for (int nt = 0; nt < 4; ++nt) {
      const bf16_t* q = Wt + (nt * 16 + l16) * K + kk + half * 8;
      v8bf b0 = *(const v8bf*)q;
      v8bf b1 = *(const v8bf*)(q + 16);
      bfr[nt] = __builtin_shufflevector(b0, b1, 0,1,2,3,4,5,6,7,8,9,10,11,12,13,14,15);
    }
    #pragma unroll
    for (int mt = 0; mt < 2; ++mt)
      #pragma unroll
      for (int nt = 0; nt < 4; ++nt)
        acc[mt][nt] = __builtin_amdgcn_wmma_f32_16x16x32_bf16(
            false, afr[mt], false, bfr[nt], (short)0, acc[mt][nt], false, false);
  }

  // ---- Store Hp (transposed, bf16) + reduce e_src / e_dst per row.
  #pragma unroll
  for (int mt = 0; mt < 2; ++mt) {
    #pragma unroll
    for (int v = 0; v < 8; ++v) {
      int row = rowbase + mt * 16 + v + 8 * half;
      float ps = 0.f, pd = 0.f;
      #pragma unroll
      for (int nt = 0; nt < 4; ++nt) {
        float val = acc[mt][nt][v];
        HpT[(nt * 16 + l16) * N + row] = (bf16_t)val;
        ps += val * asrc_l[nt];
        pd += val * adst_l[nt];
      }
      ps += __shfl_xor(ps, 1, 32); ps += __shfl_xor(ps, 2, 32);
      ps += __shfl_xor(ps, 4, 32); ps += __shfl_xor(ps, 8, 32);
      pd += __shfl_xor(pd, 1, 32); pd += __shfl_xor(pd, 2, 32);
      pd += __shfl_xor(pd, 4, 32); pd += __shfl_xor(pd, 8, 32);
      if (l16 == 0) { esrc[row] = ps; edst[row] = pd; }
    }
  }
  __syncthreads();

  // ---- Global max of e_dst (leaky is monotone => rowmax = leaky(e_src + maxdst)).
  float md = -3.402823466e38f;
  for (int i = lane; i < N; i += 32) md = fmaxf(md, edst[i]);
  #pragma unroll
  for (int off = 16; off >= 1; off >>= 1) md = fmaxf(md, __shfl_xor(md, off, 32));

  // ---- GEMM2: out = exp(leaky(e)-rowmax) @ Hp ; normalization deferred.
  v8f oacc[2][4];
  #pragma unroll
  for (int mt = 0; mt < 2; ++mt)
    #pragma unroll
    for (int nt = 0; nt < 4; ++nt)
      #pragma unroll
      for (int i = 0; i < 8; ++i) oacc[mt][nt][i] = 0.f;

  float es_l[2], rmax[2], rs[2];
  #pragma unroll
  for (int mt = 0; mt < 2; ++mt) {
    es_l[mt] = esrc[rowbase + mt * 16 + l16];
    rmax[mt] = leaky(es_l[mt] + md);
    rs[mt] = 0.f;
  }

  for (int kk = 0; kk < N; kk += 32) {
    const float* ep = edst + kk + half * 8;
    float4 ea = *(const float4*)(ep);
    float4 eb = *(const float4*)(ep + 4);
    float4 ec = *(const float4*)(ep + 16);
    float4 ef = *(const float4*)(ep + 20);
    float edv[16] = {ea.x,ea.y,ea.z,ea.w, eb.x,eb.y,eb.z,eb.w,
                     ec.x,ec.y,ec.z,ec.w, ef.x,ef.y,ef.z,ef.w};

    v16bf a2[2];
    #pragma unroll
    for (int mt = 0; mt < 2; ++mt) {
      v16bf a;
      float rsl = 0.f;
      #pragma unroll
      for (int e = 0; e < 16; ++e) {
        float p = __expf(leaky(es_l[mt] + edv[e]) - rmax[mt]);  // v_exp_f32, co-exec with WMMA
        rsl += p;
        a[e] = (bf16_t)p;
      }
      rs[mt] += rsl;
      a2[mt] = a;
    }

    v16bf b2[4];
    #pragma unroll
    for (int nt = 0; nt < 4; ++nt) {
      const bf16_t* q = HpT + (nt * 16 + l16) * N + kk + half * 8;
      v8bf b0 = *(const v8bf*)q;
      v8bf b1 = *(const v8bf*)(q + 16);
      b2[nt] = __builtin_shufflevector(b0, b1, 0,1,2,3,4,5,6,7,8,9,10,11,12,13,14,15);
    }

    #pragma unroll
    for (int mt = 0; mt < 2; ++mt)
      #pragma unroll
      for (int nt = 0; nt < 4; ++nt)
        oacc[mt][nt] = __builtin_amdgcn_wmma_f32_16x16x32_bf16(
            false, a2[mt], false, b2[nt], (short)0, oacc[mt][nt], false, false);
  }

  // Finish row sums across the two lane-halves (same row, disjoint K sets).
  #pragma unroll
  for (int mt = 0; mt < 2; ++mt) rs[mt] += __shfl_xor(rs[mt], 16, 32);

  // ---- Normalize, ELU, write out (b, head, n, d).
  float* outB = out + ((size_t)(b * NH + head)) * N * D;
  #pragma unroll
  for (int mt = 0; mt < 2; ++mt) {
    #pragma unroll
    for (int v = 0; v < 8; ++v) {
      int row = rowbase + mt * 16 + v + 8 * half;
      float rinv = 1.0f / __shfl(rs[mt], v + 8 * half, 32);
      #pragma unroll
      for (int nt = 0; nt < 4; ++nt) {
        float o = oacc[mt][nt][v] * rinv;
        o = o > 0.f ? o : (__expf(o) - 1.f);   // ELU (alpha = 1)
        outB[(size_t)row * D + nt * 16 + l16] = o;
      }
    }
  }
}

extern "C" void kernel_launch(void* const* d_in, const int* in_sizes, int n_in,
                              void* d_out, int out_size, void* d_ws, size_t ws_size,
                              hipStream_t stream) {
  const float* h     = (const float*)d_in[0];
  const float* W     = (const float*)d_in[1];
  const float* att_a = (const float*)d_in[2];
  float* out = (float*)d_out;
  // LDS: Wt 64*128*2 + HpT 64*512*2 + esrc/edst 2*512*4 = 86016 bytes (< 320 KB/WGP)
  constexpr size_t SMEM = (size_t)(64 * 128 + 64 * 512) * 2 + 2 * 512 * 4;
  dim3 grid(32 * 8);   // one workgroup per (batch, head)
  dim3 block(512);     // 16 wave32s
  gat_fused_kernel<<<grid, block, SMEM, stream>>>(h, W, att_a, out);
}
